// LSTMAutoencoder_80453327389187
// MI455X (gfx1250) — matrix operand order repaired and checked by reading, not compiled, not run on hardware
//
#include <hip/hip_runtime.h>
#include <hip/hip_bf16.h>
#include <math.h>

typedef _Float16 v16h __attribute__((ext_vector_type(16)));
typedef _Float16 v8h  __attribute__((ext_vector_type(8)));
typedef float    v8f  __attribute__((ext_vector_type(8)));

#define BATCH 64
#define SEQT  512

__device__ __forceinline__ float fast_sigmoid(float x) {
    return 1.f / (1.f + __expf(-x));
}
__device__ __forceinline__ float fast_tanh(float x) {
    return 2.f / (1.f + __expf(-2.f * x)) - 1.f;
}

// ---------------------------------------------------------------------------
// small utility kernels
// ---------------------------------------------------------------------------
__global__ void cvt_f16_kernel(_Float16* __restrict__ dst, const float* __restrict__ src, int n) {
    int i = blockIdx.x * 256 + threadIdx.x;
    if (i < n) dst[i] = (_Float16)src[i];
}

__global__ void bias_sum_kernel(float* __restrict__ dst, const float* __restrict__ a,
                                const float* __restrict__ b, int n) {
    int i = blockIdx.x * 256 + threadIdx.x;
    if (i < n) dst[i] = a[i] + b[i];
}

// (B,T,I) fp32 -> (T,B,I) f16  (time-major rows for the big input GEMM)
__global__ void xpose_cvt_kernel(_Float16* __restrict__ dst, const float* __restrict__ x) {
    int i = blockIdx.x * 256 + threadIdx.x;
    const int total = SEQT * BATCH * 256;
    if (i < total) {
        int c = i & 255;
        int b = (i >> 8) & 63;
        int t = i >> 14;
        dst[i] = (_Float16)x[((size_t)b * SEQT + t) * 256 + c];
    }
}

__global__ void zero_state_kernel(_Float16* __restrict__ h, float* __restrict__ c, int n) {
    int i = blockIdx.x * 256 + threadIdx.x;
    if (i < n) { h[i] = (_Float16)0.f; c[i] = 0.f; }
}

// ---------------------------------------------------------------------------
// Gpre = In @ Wih^T + (bih + bhh)   computed over all T*B rows in parallel
//   In : (T*B, K) f16 time-major      W : (N4, K) f16 row-major (== B operand)
//   G  : (T*B, N4) fp32
// Block: 256 thr = 8 waves; wave tile = 16x64 (4 wmma accums); block = 128x64
// ---------------------------------------------------------------------------
__global__ void __launch_bounds__(256)
gemm_gpre_kernel(const _Float16* __restrict__ In, const _Float16* __restrict__ W,
                 const float* __restrict__ bias, float* __restrict__ G,
                 int K, int N4, int reverse) {
    const int lane   = threadIdx.x & 31;
    const int wave   = threadIdx.x >> 5;
    const int laneHi = (lane >= 16) ? 1 : 0;
    const int m0 = blockIdx.x * 128 + wave * 16;
    const int n0 = blockIdx.y * 64;

    int mrow = m0 + (lane & 15);
    int in_row = mrow;
    if (reverse) {                       // decoder layer 0 consumes x reversed in time
        int t = mrow >> 6;               // row = t*B + b, B = 64
        int b = mrow & 63;
        in_row = (SEQT - 1 - t) * BATCH + b;
    }
    const _Float16* arow = In + (size_t)in_row * K + (laneHi ? 16 : 0);

    v8f acc[4] = {};
    for (int kt = 0; kt < K; kt += 32) {
        v8h a0 = *(const v8h*)(arow + kt);        // K = kt + [0..7]  (+16 if laneHi)
        v8h a1 = *(const v8h*)(arow + kt + 8);    // K = kt + [8..15] (+16 if laneHi)
        v16h a;
        #pragma unroll
        for (int j = 0; j < 8; ++j) { a[j] = a0[j]; a[8 + j] = a1[j]; }
        #pragma unroll
        for (int nt = 0; nt < 4; ++nt) {
            int n = n0 + nt * 16 + (lane & 15);
            v16h b = *(const v16h*)(W + (size_t)n * K + kt + (laneHi ? 16 : 0));
            acc[nt] = __builtin_amdgcn_wmma_f32_16x16x32_f16(
                false, a, false, b, (short)0, acc[nt], false, false);
        }
    }
    #pragma unroll
    for (int nt = 0; nt < 4; ++nt) {
        int n = n0 + nt * 16 + (lane & 15);
        float bv = bias[n];
        #pragma unroll
        for (int r = 0; r < 8; ++r) {
            int row = m0 + r + laneHi * 8;
            G[(size_t)row * N4 + n] = acc[nt][r] + bv;
        }
    }
}

// ---------------------------------------------------------------------------
// One recurrent timestep. Grid: (B/16, H/64). Block: 128 thr = 4 waves,
// wave gi computes gate-group gi's 16x64 tile of  h_prev @ Whh^T  -> LDS,
// then fused LSTM elementwise (coalesced Gpre add, single-exp sigmoid/tanh)
// -> c (f32, in place), h (f16 ping-pong + f16 sequence), optional fp32 out.
// Prefetches the next timestep's Gpre slice into L2 for the next launch.
// ---------------------------------------------------------------------------
__global__ void __launch_bounds__(128)
lstm_step_kernel(const float* __restrict__ Gt,          // (B, 4H) slice at t
                 const _Float16* __restrict__ Whh,      // (4H, H)
                 const _Float16* __restrict__ hprev,    // (B, H)
                 _Float16* __restrict__ hnew,           // (B, H)
                 float* __restrict__ c,                 // (B, H)
                 _Float16* __restrict__ seqout,         // (B, H) slice at t (time-major)
                 float* __restrict__ finout,            // d_out base or nullptr
                 int H, int t) {
    const int lane   = threadIdx.x & 31;
    const int gi     = threadIdx.x >> 5;          // gate: 0=i 1=f 2=g 3=o
    const int laneHi = (lane >= 16) ? 1 : 0;
    const int m0 = blockIdx.x * 16;
    const int n0 = blockIdx.y * 64;
    const int N4 = 4 * H;

    __shared__ float g4[4][16][64];               // 16 KB: raw h@Whh^T per gate

    const _Float16* arow = hprev + (size_t)(m0 + (lane & 15)) * H + (laneHi ? 16 : 0);

    v8f acc[4] = {};
    for (int kt = 0; kt < H; kt += 32) {
        v8h a0 = *(const v8h*)(arow + kt);
        v8h a1 = *(const v8h*)(arow + kt + 8);
        v16h a;
        #pragma unroll
        for (int j = 0; j < 8; ++j) { a[j] = a0[j]; a[8 + j] = a1[j]; }
        #pragma unroll
        for (int nt = 0; nt < 4; ++nt) {
            int n = gi * H + n0 + nt * 16 + (lane & 15);
            v16h b = *(const v16h*)(Whh + (size_t)n * H + kt + (laneHi ? 16 : 0));
            acc[nt] = __builtin_amdgcn_wmma_f32_16x16x32_f16(
                false, a, false, b, (short)0, acc[nt], false, false);
        }
    }
    #pragma unroll
    for (int nt = 0; nt < 4; ++nt) {
        int nl = nt * 16 + (lane & 15);
        #pragma unroll
        for (int r = 0; r < 8; ++r)
            g4[gi][r + laneHi * 8][nl] = acc[nt][r];
    }
    __syncthreads();

    for (int e = threadIdx.x; e < 1024; e += 128) {
        int m  = e >> 6;
        int nl = e & 63;
        int b = m0 + m, col = n0 + nl;
        const float* grow = Gt + (size_t)b * N4 + n0 + nl;
        float iv = g4[0][m][nl] + grow[0 * H];    // coalesced: nl contiguous in lanes
        float fv = g4[1][m][nl] + grow[1 * H];
        float gv = g4[2][m][nl] + grow[2 * H];
        float ov = g4[3][m][nl] + grow[3 * H];
        float si = fast_sigmoid(iv);
        float sf = fast_sigmoid(fv);
        float so = fast_sigmoid(ov);
        float tg = fast_tanh(gv);
        size_t idx = (size_t)b * H + col;
        float cn = sf * c[idx] + si * tg;
        float hn = so * fast_tanh(cn);
        c[idx]      = cn;
        hnew[idx]   = (_Float16)hn;
        seqout[idx] = (_Float16)hn;
        if (finout)  // d_out layout (B, T, 256)
            finout[((size_t)b * SEQT + t) * (size_t)H + col] = hn;
    }

    // warm L2 with the next step's Gpre slice (consumed by the next launch)
    if (t + 1 < SEQT) {
        const float* gnext = Gt + (size_t)BATCH * N4
                           + (size_t)(m0 + (threadIdx.x & 15)) * N4 + n0;
        #pragma unroll
        for (int g = 0; g < 4; ++g)
            __builtin_prefetch(gnext + g * H + (threadIdx.x >> 4) * 8, 0, 3);
    }
}

// ---------------------------------------------------------------------------
// host orchestration
// ---------------------------------------------------------------------------
extern "C" void kernel_launch(void* const* d_in, const int* in_sizes, int n_in,
                              void* d_out, int out_size, void* d_ws, size_t ws_size,
                              hipStream_t stream) {
    (void)in_sizes; (void)n_in; (void)out_size; (void)ws_size;
    const float* x = (const float*)d_in[0];

    struct LayerDef { int in, H, wih, whh, b1, b2; };
    const LayerDef L[5] = {
        {256, 512,  1,  2,  3,  4},   // enc0
        {512, 512,  5,  6,  7,  8},   // enc1
        {256, 512,  9, 10, 11, 12},   // dec0 (reversed input, state carried from enc1)
        {512, 512, 13, 14, 15, 16},   // dec1
        {512, 256, 17, 18, 19, 20},   // dec2 -> d_out
    };

    // ---- carve workspace (all offsets 256B aligned) ----
    char* ws = (char*)d_ws;
    size_t off = 0;
    auto take = [&](size_t bytes) { void* p = ws + off; off += (bytes + 255) & ~(size_t)255; return p; };

    _Float16* wih16[5]; _Float16* whh16[5]; float* bias32[5];
    for (int l = 0; l < 5; ++l) {
        wih16[l]  = (_Float16*)take((size_t)4 * L[l].H * L[l].in * sizeof(_Float16));
        whh16[l]  = (_Float16*)take((size_t)4 * L[l].H * L[l].H  * sizeof(_Float16));
        bias32[l] = (float*)   take((size_t)4 * L[l].H * sizeof(float));
    }
    _Float16* x16   = (_Float16*)take((size_t)SEQT * BATCH * 256 * sizeof(_Float16));
    _Float16* seq0  = (_Float16*)take((size_t)SEQT * BATCH * 512 * sizeof(_Float16));
    _Float16* seq1  = (_Float16*)take((size_t)SEQT * BATCH * 512 * sizeof(_Float16));
    float*    gpre  = (float*)   take((size_t)SEQT * BATCH * 2048 * sizeof(float));
    _Float16* hbuf0 = (_Float16*)take((size_t)BATCH * 512 * sizeof(_Float16));
    _Float16* hbuf1 = (_Float16*)take((size_t)BATCH * 512 * sizeof(_Float16));
    float*    cbuf  = (float*)   take((size_t)BATCH * 512 * sizeof(float));

    // ---- convert weights to f16, fold biases ----
    for (int l = 0; l < 5; ++l) {
        int nwih = 4 * L[l].H * L[l].in;
        int nwhh = 4 * L[l].H * L[l].H;
        int nb   = 4 * L[l].H;
        cvt_f16_kernel<<<(nwih + 255) / 256, 256, 0, stream>>>(wih16[l], (const float*)d_in[L[l].wih], nwih);
        cvt_f16_kernel<<<(nwhh + 255) / 256, 256, 0, stream>>>(whh16[l], (const float*)d_in[L[l].whh], nwhh);
        bias_sum_kernel<<<(nb + 255) / 256, 256, 0, stream>>>(bias32[l], (const float*)d_in[L[l].b1],
                                                              (const float*)d_in[L[l].b2], nb);
    }
    {
        int nx = SEQT * BATCH * 256;
        xpose_cvt_kernel<<<(nx + 255) / 256, 256, 0, stream>>>(x16, x);
    }

    // ---- one LSTM layer pass: big input GEMM, then T sequential steps ----
    auto run_layer = [&](int l, const _Float16* inseq, _Float16* outseq,
                         bool reverse, bool zero_init, float* finout) {
        const int H = L[l].H, K = L[l].in, N4 = 4 * H;
        dim3 gg(SEQT * BATCH / 128, N4 / 64);
        gemm_gpre_kernel<<<gg, 256, 0, stream>>>(inseq, wih16[l], bias32[l], gpre, K, N4, reverse ? 1 : 0);
        if (zero_init) {
            int n = BATCH * H;
            zero_state_kernel<<<(n + 255) / 256, 256, 0, stream>>>(hbuf0, cbuf, n);
        }
        for (int t = 0; t < SEQT; ++t) {
            const _Float16* hp = (t & 1) ? hbuf1 : hbuf0;
            _Float16*       hn = (t & 1) ? hbuf0 : hbuf1;
            lstm_step_kernel<<<dim3(BATCH / 16, H / 64), 128, 0, stream>>>(
                gpre + (size_t)t * BATCH * N4, whh16[l], hp, hn, cbuf,
                outseq + (size_t)t * BATCH * H, finout, H, t);
        }
        // T even -> final state ends up in hbuf0 / cbuf (consumed by dec0 as h0/c0)
    };

    run_layer(0, x16,  seq0, false, true,  nullptr);           // enc0
    run_layer(1, seq0, seq1, false, true,  nullptr);           // enc1
    run_layer(2, x16,  seq0, true,  false, nullptr);           // dec0: reversed x, enc1 state
    run_layer(3, seq0, seq1, false, true,  nullptr);           // dec1
    run_layer(4, seq1, seq0, false, true,  (float*)d_out);     // dec2 -> (B,T,256) fp32
}
